// SelfAttention_22024592294626
// MI455X (gfx1250) — compile-verified, matrix operand order checked by reading, not run on hardware
//
#include <hip/hip_runtime.h>
#include <hip/hip_bf16.h>

// CDNA5 WMMA types
typedef __attribute__((ext_vector_type(16))) __bf16 v16bf;
typedef __attribute__((ext_vector_type(8)))  __bf16 bf16x8;
typedef __attribute__((ext_vector_type(2)))  __bf16 bf16x2;
typedef __attribute__((ext_vector_type(8)))  float  v8f;

#define S_LEN 2048
#define E_DIM 1024
#define H_NUM 16
#define D_DIM 64
#define B_NUM 2
#define M_TOT 4096

// A/B bf16 fragment (16x32) from row-major bf16 storage: elements 0..7 are
// k = half16*8 + {0..7}, elements 8..15 are k = 16 + half16*8 + {0..7}
// -> exactly two contiguous b128 loads per fragment.
static __device__ __forceinline__ v16bf load_frag(const __bf16* row, int koff) {
    const bf16x8 lo = *(const bf16x8*)(row + koff);
    const bf16x8 hi = *(const bf16x8*)(row + koff + 16);
    return __builtin_shufflevector(lo, hi, 0,1,2,3,4,5,6,7,8,9,10,11,12,13,14,15);
}

static __device__ __forceinline__ v8f wmma_bf16(v16bf a, v16bf b, v8f c) {
    return __builtin_amdgcn_wmma_f32_16x16x32_bf16(false, a, false, b,
                                                   (short)0, c, false, false);
}

// -------------------------------------------------------------------------
// Kernel 0: one-pass f32 -> bf16 conversion (x and the four weights), so
// the GEMM hot loops carry zero cvt VALU work and use b128 fragment loads.
// -------------------------------------------------------------------------
__launch_bounds__(256)
__global__ void cvt_kernel(const float* __restrict__ src, __bf16* __restrict__ dst)
{
    const size_t i = ((size_t)blockIdx.x * 256 + threadIdx.x) * 8;
    const float4 u = *(const float4*)(src + i);
    const float4 v = *(const float4*)(src + i + 4);
    bf16x8 o;
    o[0] = (__bf16)u.x; o[1] = (__bf16)u.y; o[2] = (__bf16)u.z; o[3] = (__bf16)u.w;
    o[4] = (__bf16)v.x; o[5] = (__bf16)v.y; o[6] = (__bf16)v.z; o[7] = (__bf16)v.w;
    *(bf16x8*)(dst + i) = o;
}

// -------------------------------------------------------------------------
// Kernel 1: QKV projection, y = x @ W.T + b.  Register-blocked: each wave
// computes a 32x32 tile (2 A-frags x 2 B-frags -> 4 WMMAs per K=32 step).
// Q,K stored bf16 [B][H][S][D]; V stored bf16 [B][H][D][S] (transposed).
// -------------------------------------------------------------------------
__launch_bounds__(256)
__global__ void qkv_proj_kernel(const __bf16* __restrict__ xb,
                                const __bf16* __restrict__ wqb,
                                const __bf16* __restrict__ wkb,
                                const __bf16* __restrict__ wvb,
                                const float* __restrict__ bq,
                                const float* __restrict__ bk,
                                const float* __restrict__ bv,
                                __bf16* __restrict__ qbuf,
                                __bf16* __restrict__ kbuf,
                                __bf16* __restrict__ vtbuf)
{
    const int lane   = threadIdx.x & 31;
    const int wave   = threadIdx.x >> 5;
    const int half16 = (lane >> 4) & 1;
    const int ln     = lane & 15;
    const int kbase  = half16 * 8;

    const int which = blockIdx.z;                 // 0=Q 1=K 2=V
    const __bf16* Wb  = (which == 0) ? wqb : (which == 1) ? wkb : wvb;
    const float* bias = (which == 0) ? bq  : (which == 1) ? bk  : bv;

    // block tile 128(M) x 64(N); wave tile 32x32
    const int m_base = blockIdx.y * 128 + (wave & 3) * 32;
    const int n_base = blockIdx.x * 64  + (wave >> 2) * 32;

    const __bf16* a0 = xb + (size_t)(m_base + ln) * E_DIM;
    const __bf16* a1 = a0 + (size_t)16 * E_DIM;
    const __bf16* b0 = Wb + (size_t)(n_base + ln) * E_DIM;
    const __bf16* b1 = b0 + (size_t)16 * E_DIM;

    v8f c00, c01, c10, c11;
#pragma unroll
    for (int r = 0; r < 8; ++r) { c00[r] = 0.f; c01[r] = 0.f; c10[r] = 0.f; c11[r] = 0.f; }

    for (int k0 = 0; k0 < E_DIM; k0 += 32) {
        __builtin_prefetch(a0 + k0 + 128, 0, 3);   // global_prefetch_b8, next steps
        __builtin_prefetch(b0 + k0 + 128, 0, 3);
        const v16bf fa0 = load_frag(a0, k0 + kbase);
        const v16bf fa1 = load_frag(a1, k0 + kbase);
        const v16bf fb0 = load_frag(b0, k0 + kbase);
        const v16bf fb1 = load_frag(b1, k0 + kbase);
        c00 = wmma_bf16(fa0, fb0, c00);
        c01 = wmma_bf16(fa0, fb1, c01);
        c10 = wmma_bf16(fa1, fb0, c10);
        c11 = wmma_bf16(fa1, fb1, c11);
    }

    const float bias0 = bias[n_base + ln];
    const float bias1 = bias[n_base + 16 + ln];

#pragma unroll
    for (int fi = 0; fi < 2; ++fi) {
#pragma unroll
        for (int fj = 0; fj < 2; ++fj) {
            const v8f  cc = (fi == 0) ? (fj == 0 ? c00 : c01)
                                      : (fj == 0 ? c10 : c11);
            const float bb = fj ? bias1 : bias0;
            const int n = n_base + fj * 16 + ln;
            const int h = n >> 6, d = n & (D_DIM - 1);
            if (which == 2) {
                // V^T [B][H][D][S]: consecutive r -> consecutive s, pack 2/dword
#pragma unroll
                for (int r = 0; r < 8; r += 2) {
                    const int m  = m_base + fi * 16 + r + half16 * 8;
                    const int bi = m >> 11, s = m & (S_LEN - 1);
                    bf16x2 p;
                    p[0] = (__bf16)(cc[r] + bb);
                    p[1] = (__bf16)(cc[r + 1] + bb);
                    *(bf16x2*)(vtbuf + ((size_t)(bi * H_NUM + h) * D_DIM + d) * S_LEN + s) = p;
                }
            } else {
                __bf16* dst = (which == 0) ? qbuf : kbuf;
#pragma unroll
                for (int r = 0; r < 8; ++r) {
                    const int m  = m_base + fi * 16 + r + half16 * 8;
                    const int bi = m >> 11, s = m & (S_LEN - 1);
                    dst[((size_t)(bi * H_NUM + h) * S_LEN + s) * D_DIM + d] =
                        (__bf16)(cc[r] + bb);
                }
            }
        }
    }
}

// -------------------------------------------------------------------------
// Kernel 2: attention; softmax over the HEADS axis (faithful quirk).
// Block = 512 threads = 16 waves; wave h owns head h for one 16-query tile.
// Per 32-key step: 4 score WMMAs -> LDS [(q,k)][h] -> cross-head max/sumexp
// (one (q,k) per thread) -> attn rebuilt in A-layout -> 4 attn@V WMMAs.
// -------------------------------------------------------------------------
__launch_bounds__(512)
__global__ void attn_kernel(const __bf16* __restrict__ qbuf,
                            const __bf16* __restrict__ kbuf,
                            const __bf16* __restrict__ vtbuf,
                            __bf16* __restrict__ aobuf)
{
    __shared__ float lds_s[16 * 32 * 17];  // [(q*32+k)*17 + h], padded stride
    __shared__ float lds_m[512];
    __shared__ float lds_r[512];

    const int lane   = threadIdx.x & 31;
    const int h      = threadIdx.x >> 5;   // wave index == head
    const int half16 = (lane >> 4) & 1;
    const int ln     = lane & 15;
    const int kbase  = half16 * 8;

    const int b  = blockIdx.y;
    const int q0 = blockIdx.x * 16;

    const __bf16* qhead = qbuf  + (size_t)(b * H_NUM + h) * S_LEN * D_DIM;
    const __bf16* khead = kbuf  + (size_t)(b * H_NUM + h) * S_LEN * D_DIM;
    const __bf16* vhead = vtbuf + (size_t)(b * H_NUM + h) * D_DIM * S_LEN;

    // Q A-fragments (d = 0..31, 32..63), resident the whole kernel
    const __bf16* qrow = qhead + (size_t)(q0 + ln) * D_DIM;
    const v16bf aq0 = load_frag(qrow, kbase);
    const v16bf aq1 = load_frag(qrow, 32 + kbase);

    v8f o[4];
#pragma unroll
    for (int cc = 0; cc < 4; ++cc)
#pragma unroll
        for (int r = 0; r < 8; ++r) o[cc][r] = 0.f;

    for (int k0 = 0; k0 < S_LEN; k0 += 32) {
        const __bf16* kr0 = khead + (size_t)(k0 + ln) * D_DIM;
        const __bf16* kr1 = kr0 + (size_t)16 * D_DIM;
        __builtin_prefetch(kr0 + 32 * D_DIM, 0, 3);

        const v16bf b00 = load_frag(kr0, kbase);
        const v16bf b01 = load_frag(kr0, 32 + kbase);
        const v16bf b10 = load_frag(kr1, kbase);
        const v16bf b11 = load_frag(kr1, 32 + kbase);

        v8f s0, s1;
#pragma unroll
        for (int r = 0; r < 8; ++r) { s0[r] = 0.f; s1[r] = 0.f; }
        s0 = wmma_bf16(aq0, b00, s0);
        s0 = wmma_bf16(aq1, b01, s0);
        s1 = wmma_bf16(aq0, b10, s1);
        s1 = wmma_bf16(aq1, b11, s1);

        const float scale = 0.125f;  // 1/sqrt(64)
#pragma unroll
        for (int r = 0; r < 8; ++r) {
            const int ql = r + half16 * 8;
            lds_s[(ql * 32 + ln)      * 17 + h] = s0[r] * scale;
            lds_s[(ql * 32 + 16 + ln) * 17 + h] = s1[r] * scale;
        }
        __syncthreads();

        {   // cross-head reduction: 512 (q,k) pairs, one per thread
            const int pair = threadIdx.x;
            const float* col = &lds_s[pair * 17];
            float mx = col[0];
#pragma unroll
            for (int hh = 1; hh < 16; ++hh) mx = fmaxf(mx, col[hh]);
            float se = 0.f;
#pragma unroll
            for (int hh = 0; hh < 16; ++hh) se += __expf(col[hh] - mx);
            lds_m[pair] = mx;
            lds_r[pair] = 1.0f / se;
        }
        __syncthreads();

        // attn in A-fragment layout (16 q x 32 keys), normalized over heads
        v16bf at;
#pragma unroll
        for (int j = 0; j < 8; ++j) {
            const int kk = (j >> 2) * 16 + half16 * 8 + (j & 3) * 2;
#pragma unroll
            for (int t = 0; t < 2; ++t) {
                const int pair = ln * 32 + kk + t;   // q = ln, key = kk+t
                const float sv = lds_s[pair * 17 + h];
                const float av = __expf(sv - lds_m[pair]) * lds_r[pair];
                at[2 * j + t] = (__bf16)av;
            }
        }

        // out += attn(16x32) @ V(32keys x 64d), 4 N-chunks of 16
#pragma unroll
        for (int cc = 0; cc < 4; ++cc) {
            const __bf16* vrow = vhead + (size_t)(cc * 16 + ln) * S_LEN + k0;
            const v16bf bv = load_frag(vrow, kbase);
            o[cc] = wmma_bf16(at, bv, o[cc]);
        }
        __syncthreads();  // protect lds_s before next k-step
    }

    // store attn-out bf16 [B*S][E], col = h*64 + d
#pragma unroll
    for (int cc = 0; cc < 4; ++cc)
#pragma unroll
        for (int r = 0; r < 8; ++r) {
            const int q   = q0 + r + half16 * 8;
            const int col = h * D_DIM + cc * 16 + ln;
            aobuf[(size_t)(b * S_LEN + q) * E_DIM + col] = (__bf16)o[cc][r];
        }
}

// -------------------------------------------------------------------------
// Kernel 3: output projection.  out = attn_out @ Wo.T + bo (f32 output).
// Same 32x32 per-wave register blocking as kernel 1.
// -------------------------------------------------------------------------
__launch_bounds__(256)
__global__ void oproj_kernel(const __bf16* __restrict__ ao,
                             const __bf16* __restrict__ wob,
                             const float* __restrict__ bo,
                             float* __restrict__ out)
{
    const int lane   = threadIdx.x & 31;
    const int wave   = threadIdx.x >> 5;
    const int half16 = (lane >> 4) & 1;
    const int ln     = lane & 15;
    const int kbase  = half16 * 8;

    const int m_base = blockIdx.y * 128 + (wave & 3) * 32;
    const int n_base = blockIdx.x * 64  + (wave >> 2) * 32;

    const __bf16* a0 = ao  + (size_t)(m_base + ln) * E_DIM;
    const __bf16* a1 = a0  + (size_t)16 * E_DIM;
    const __bf16* b0 = wob + (size_t)(n_base + ln) * E_DIM;
    const __bf16* b1 = b0  + (size_t)16 * E_DIM;

    v8f c00, c01, c10, c11;
#pragma unroll
    for (int r = 0; r < 8; ++r) { c00[r] = 0.f; c01[r] = 0.f; c10[r] = 0.f; c11[r] = 0.f; }

    for (int k0 = 0; k0 < E_DIM; k0 += 32) {
        __builtin_prefetch(a0 + k0 + 128, 0, 3);
        __builtin_prefetch(b0 + k0 + 128, 0, 3);
        const v16bf fa0 = load_frag(a0, k0 + kbase);
        const v16bf fa1 = load_frag(a1, k0 + kbase);
        const v16bf fb0 = load_frag(b0, k0 + kbase);
        const v16bf fb1 = load_frag(b1, k0 + kbase);
        c00 = wmma_bf16(fa0, fb0, c00);
        c01 = wmma_bf16(fa0, fb1, c01);
        c10 = wmma_bf16(fa1, fb0, c10);
        c11 = wmma_bf16(fa1, fb1, c11);
    }

    const float bias0 = bo[n_base + ln];
    const float bias1 = bo[n_base + 16 + ln];

#pragma unroll
    for (int fi = 0; fi < 2; ++fi)
#pragma unroll
        for (int fj = 0; fj < 2; ++fj) {
            const v8f  cc = (fi == 0) ? (fj == 0 ? c00 : c01)
                                      : (fj == 0 ? c10 : c11);
            const float bb = fj ? bias1 : bias0;
            const int n = n_base + fj * 16 + ln;
#pragma unroll
            for (int r = 0; r < 8; ++r) {
                const int m = m_base + fi * 16 + r + half16 * 8;
                out[(size_t)m * E_DIM + n] = cc[r] + bb;
            }
        }
}

// -------------------------------------------------------------------------
extern "C" void kernel_launch(void* const* d_in, const int* in_sizes, int n_in,
                              void* d_out, int out_size, void* d_ws, size_t ws_size,
                              hipStream_t stream)
{
    const float* x  = (const float*)d_in[0];
    const float* Wq = (const float*)d_in[1];
    const float* bq = (const float*)d_in[2];
    const float* Wk = (const float*)d_in[3];
    const float* bk = (const float*)d_in[4];
    const float* Wv = (const float*)d_in[5];
    const float* bv = (const float*)d_in[6];
    const float* Wo = (const float*)d_in[7];
    const float* bo = (const float*)d_in[8];
    float* out = (float*)d_out;

    // Workspace (bf16): Q 8MiB | K 8MiB | V^T 8MiB | attn-out 8MiB |
    //                   x 8MiB | Wq 2 | Wk 2 | Wv 2 | Wo 2   = 48 MiB
    const size_t MB = 1024 * 1024;
    char* ws = (char*)d_ws;
    __bf16* qbuf  = (__bf16*)(ws);
    __bf16* kbuf  = (__bf16*)(ws + 8 * MB);
    __bf16* vtbuf = (__bf16*)(ws + 16 * MB);
    __bf16* aobuf = (__bf16*)(ws + 24 * MB);
    __bf16* xb    = (__bf16*)(ws + 32 * MB);
    __bf16* wqb   = (__bf16*)(ws + 40 * MB);
    __bf16* wkb   = (__bf16*)(ws + 42 * MB);
    __bf16* wvb   = (__bf16*)(ws + 44 * MB);
    __bf16* wob   = (__bf16*)(ws + 46 * MB);
    (void)in_sizes; (void)n_in; (void)out_size; (void)ws_size;

    // K0: f32 -> bf16 one-pass conversions (8 elems/thread)
    cvt_kernel<<<(M_TOT * E_DIM) / (8 * 256), 256, 0, stream>>>(x, xb);
    cvt_kernel<<<(E_DIM * E_DIM) / (8 * 256), 256, 0, stream>>>(Wq, wqb);
    cvt_kernel<<<(E_DIM * E_DIM) / (8 * 256), 256, 0, stream>>>(Wk, wkb);
    cvt_kernel<<<(E_DIM * E_DIM) / (8 * 256), 256, 0, stream>>>(Wv, wvb);
    cvt_kernel<<<(E_DIM * E_DIM) / (8 * 256), 256, 0, stream>>>(Wo, wob);

    // K1: QKV projection (block tile 128x64, wave tile 32x32)
    dim3 g1(E_DIM / 64, M_TOT / 128, 3);
    qkv_proj_kernel<<<g1, 256, 0, stream>>>(xb, wqb, wkb, wvb, bq, bk, bv,
                                            qbuf, kbuf, vtbuf);

    // K2: attention, one block per (q-tile, batch); 16 waves = 16 heads
    dim3 g2(S_LEN / 16, B_NUM);
    attn_kernel<<<g2, 512, 0, stream>>>(qbuf, kbuf, vtbuf, aobuf);

    // K3: output projection
    dim3 g3(E_DIM / 64, M_TOT / 128);
    oproj_kernel<<<g3, 256, 0, stream>>>(aobuf, Wo ? wob : wob, bo, out);
}
